// MaskedAutoEncoder_40913858462145
// MI455X (gfx1250) — matrix operand order repaired and checked by reading, not compiled, not run on hardware
//
#include <hip/hip_runtime.h>
#include <stdint.h>

typedef __attribute__((ext_vector_type(16))) __bf16 v16bf;
typedef __attribute__((ext_vector_type(8)))  float  v8f;

struct __align__(16) U4 { unsigned int x, y, z, w; };
union Frag { U4 q[2]; v16bf v; };

#define BATCH 64
#define LPAT  196
#define KEEPN 49
#define RPB   50            /* 49 kept + 1 cls row per batch */
#define MROWS (BATCH*RPB)   /* 3200 */
#define DENC  768
#define DDEC  512
#define PDIM  768
#define MASKM 256           /* 196 mask rows padded to 256 */
#define LDSA_STRIDE 40      /* ushorts per 32-K row, padded vs 32 to dodge bank conflicts */

__device__ __forceinline__ unsigned short f2bf(float f) {
  unsigned int u = __float_as_uint(f);
  u += 0x7FFFu + ((u >> 16) & 1u);
  return (unsigned short)(u >> 16);
}

/* ---------- prep: f32 -> bf16, transposed to [N x K] for the GEMM B side --- */
__global__ void mae_cvt_t_bf16(const float* __restrict__ src,
                               unsigned short* __restrict__ dst, int K, int N) {
  int i = blockIdx.x * blockDim.x + threadIdx.x;   // over K*N, coalesced read
  if (i >= K * N) return;
  int k = i / N, n = i % N;
  dst[(size_t)n * K + k] = f2bf(src[i]);
}

/* ---------- prep: maskrow[256x512] = bf16(mask_token + dec_pos[l]) --------- */
__global__ void mae_maskrow(const float* __restrict__ mask_tok,
                            const float* __restrict__ dec_pos,
                            unsigned short* __restrict__ maskrow) {
  int i = blockIdx.x * blockDim.x + threadIdx.x;   // over 256*512
  int row = i >> 9, n = i & 511;
  float v = (row < LPAT) ? (mask_tok[n] + dec_pos[row * DDEC + n]) : 0.0f;
  maskrow[i] = f2bf(v);
}

/* ---------- prep: per-batch index maps + mask output + cls rows of E ------- */
__global__ void mae_indices(const int* __restrict__ ids_shuffle,
                            const float* __restrict__ cls_tok,
                            int* __restrict__ keep_l,     // [3200] row -> patch l (-1 = cls row)
                            int* __restrict__ pos,        // [3200] row -> output position l
                            int* __restrict__ sel,        // [64*196] -> m or -1
                            float* __restrict__ mask_out, // [64*196]
                            unsigned short* __restrict__ E) {
  int b = blockIdx.x, t = threadIdx.x;
  if (t < LPAT) { sel[b * LPAT + t] = -1; mask_out[b * LPAT + t] = 1.0f; }
  __syncthreads();
  if (t < LPAT) {
    int s = ids_shuffle[b * LPAT + t];
    if (t < KEEPN) { keep_l[b * RPB + t] = s; mask_out[b * LPAT + s] = 0.0f; }
    if (t == KEEPN) keep_l[b * RPB + KEEPN] = -1;
    if (t >= 146) { int m = t - 146; pos[b * RPB + m] = s; sel[b * LPAT + s] = m; }
  }
  for (int i = t; i < DENC; i += blockDim.x)
    E[(size_t)(b * RPB + KEEPN) * DENC + i] = f2bf(cls_tok[i]);
}

/* ---------- prep: gather kept patches (patchify order p,q,c) -> bf16 A1 ---- */
__global__ void mae_gather(const float* __restrict__ images,
                           const int* __restrict__ ids_shuffle,
                           unsigned short* __restrict__ A1) {
  int row = blockIdx.x;              // 0..3199 = b*50+k
  int b = row / RPB, k = row % RPB;
  int t = threadIdx.x;               // 256 threads, 3 elems each
  if (k == KEEPN) {                  // pad row (cls slot in A1) -> zeros
    A1[(size_t)row * PDIM + 3 * t + 0] = 0;
    A1[(size_t)row * PDIM + 3 * t + 1] = 0;
    A1[(size_t)row * PDIM + 3 * t + 2] = 0;
    return;
  }
  int l = ids_shuffle[b * LPAT + k];
  int pr = l / 14, pc = l % 14;
  int pi = t >> 4, qi = t & 15;
#pragma unroll
  for (int c = 0; c < 3; ++c) {
    size_t gi = ((size_t)(b * 3 + c) * 224 + (pr * 16 + pi)) * 224 + pc * 16 + qi;
    A1[(size_t)row * PDIM + t * 3 + c] = f2bf(images[gi]);
  }
}

/* ---------- tiled bf16 WMMA GEMM: C = A(MxK) @ Wt(NxK)^T, fused epilogues -- */
/* block tile 128x64, 8 waves x (16M x 64N), K-step 32, swizzled LDS tiles so */
/* every fragment is two contiguous ds_load_b128s; global loads pipelined.    */
__global__ __launch_bounds__(256)
void mae_gemm(const unsigned short* __restrict__ A,
              const unsigned short* __restrict__ Bw,   /* N x K (pre-transposed) */
              int M, int N, int K, int mode,
              const float* __restrict__ bias,
              const int* __restrict__ rowmap,
              const float* __restrict__ enc_pos,
              unsigned short* __restrict__ outbf,
              float* __restrict__ outf) {
  __shared__ unsigned short sA[128 * LDSA_STRIDE];   /* 10240 B */
  __shared__ unsigned short sB[64 * LDSA_STRIDE];    /*  5120 B */
  int t = threadIdx.x, lane = t & 31, wave = t >> 5;
  int m0 = blockIdx.y * 128, n0 = blockIdx.x * 64;

  /* staging geometry: 16B chunk cc covers K = cc*8..cc*8+7 of one tile row.  */
  int r = t >> 2, cc = t & 3;
  /* A swizzle: chunk cc -> lane-half h=cc&1, frag-half = cc>>1               */
  int saOff0 = r * LDSA_STRIDE + (cc & 1) * 16 + (cc >> 1) * 8;
  int saOff1 = saOff0 + 64 * LDSA_STRIDE;
  /* B swizzle: chunk cc -> lane-half h=cc&1, frag-half = cc>>1               */
  int sbOff  = r * LDSA_STRIDE + (cc & 1) * 16 + (cc >> 1) * 8;

  const unsigned short* gA0 = A  + (size_t)(m0 + r)      * K + cc * 8;
  const unsigned short* gA1 = gA0 + (size_t)64 * K;
  const unsigned short* gB  = Bw + (size_t)(n0 + r)      * K + cc * 8;

  v8f acc[4];
#pragma unroll
  for (int j = 0; j < 4; ++j)
#pragma unroll
    for (int q = 0; q < 8; ++q) acc[j][q] = 0.0f;

  U4 ra0 = *(const U4*)gA0;
  U4 ra1 = *(const U4*)gA1;
  U4 rb  = *(const U4*)gB;

  int mrow = (wave << 4) + (lane & 15);
  int half = lane >> 4;                      /* A: lane-half h ; B: kh */
  const unsigned short* pa = &sA[mrow * LDSA_STRIDE + half * 16];

  for (int k0 = 0; k0 < K; k0 += 32) {
    *(U4*)&sA[saOff0] = ra0;
    *(U4*)&sA[saOff1] = ra1;
    *(U4*)&sB[sbOff]  = rb;
    __syncthreads();

    if (k0 + 32 < K) {                       /* pipeline next tile's globals */
      gA0 += 32; gA1 += 32; gB += 32;
      ra0 = *(const U4*)gA0;
      ra1 = *(const U4*)gA1;
      rb  = *(const U4*)gB;
    }

    Frag af;
    af.q[0] = *(const U4*)(pa);
    af.q[1] = *(const U4*)(pa + 8);
#pragma unroll
    for (int j = 0; j < 4; ++j) {
      Frag bfr;
      const unsigned short* pb =
          &sB[(j * 16 + (lane & 15)) * LDSA_STRIDE + half * 16];
      bfr.q[0] = *(const U4*)(pb);
      bfr.q[1] = *(const U4*)(pb + 8);
      acc[j] = __builtin_amdgcn_wmma_f32_16x16x32_bf16(
          false, af.v, false, bfr.v, (short)0, acc[j], false, false);
    }
    __syncthreads();
  }

  /* epilogue: C layout — VGPR q: lanes0-15 M=q, lanes16-31 M=q+8 */
  int ccol = lane & 15;
#pragma unroll
  for (int j = 0; j < 4; ++j) {
#pragma unroll
    for (int q = 0; q < 8; ++q) {
      int gm = m0 + (wave << 4) + q + 8 * half;
      int gn = n0 + j * 16 + ccol;
      float val = acc[j][q];
      if (mode == 0) {                               // -> E (bf16), +b_enc+enc_pos[l]
        int l = rowmap[gm];
        if (l >= 0) {
          val += bias[gn] + enc_pos[(size_t)l * DENC + gn];
          outbf[(size_t)gm * DENC + gn] = f2bf(val);
        }
      } else if (mode == 1 || mode == 2) {           // -> LP / maskpred (f32), +bias
        outf[(size_t)gm * N + gn] = val + bias[gn];
      } else {                                       // scatter kept preds into d_out
        int lo = rowmap[gm];
        int b = gm / RPB;
        outf[((size_t)b * LPAT + lo) * PDIM + gn] = val + bias[gn];
      }
    }
  }
}

/* ---------- build decoder input rows for kept positions -------------------- */
__global__ void mae_build_kd(const float* __restrict__ LP,
                             const float* __restrict__ dec_pos,
                             const int* __restrict__ pos,
                             unsigned short* __restrict__ Kd) {
  int row = blockIdx.x;                 // b*50+m
  int b = row / RPB, m = row % RPB;
  int src = b * RPB + (m + 1 < 50 ? m + 1 : 49);   // min(m+1,49): clip semantics
  int l = pos[row];
  for (int n = threadIdx.x; n < DDEC; n += blockDim.x)
    Kd[(size_t)row * DDEC + n] = f2bf(LP[(size_t)src * DDEC + n] + dec_pos[l * DDEC + n]);
}

/* ---------- broadcast precomputed masked-row predictions ------------------- */
__global__ void mae_fill_masked(const int* __restrict__ sel,
                                const float* __restrict__ maskpred,
                                float* __restrict__ pred) {
  int idx = blockIdx.x;                 // b*196+l
  if (sel[idx] >= 0) return;            // kept position: written by scatter GEMM
  int l = idx % LPAT;
  for (int n = threadIdx.x; n < PDIM; n += blockDim.x)
    pred[(size_t)idx * PDIM + n] = maskpred[(size_t)l * PDIM + n];
}

extern "C" void kernel_launch(void* const* d_in, const int* in_sizes, int n_in,
                              void* d_out, int out_size, void* d_ws, size_t ws_size,
                              hipStream_t stream) {
  const float* images   = (const float*)d_in[0];
  const int*   ids      = (const int*)  d_in[1];
  const float* W_enc    = (const float*)d_in[2];
  const float* b_enc    = (const float*)d_in[3];
  const float* W_dec    = (const float*)d_in[4];
  const float* b_dec    = (const float*)d_in[5];
  const float* W_out    = (const float*)d_in[6];
  const float* b_out    = (const float*)d_in[7];
  const float* cls_tok  = (const float*)d_in[8];
  const float* enc_pos  = (const float*)d_in[9];
  const float* dec_pos  = (const float*)d_in[10];
  const float* mask_tok = (const float*)d_in[11];

  float* pred     = (float*)d_out;
  float* mask_out = pred + (size_t)BATCH * LPAT * PDIM;

  char* ws = (char*)d_ws;
  auto carve = [&](size_t bytes) {
    char* p = ws; ws += (bytes + 255) & ~(size_t)255; return p;
  };
  unsigned short* W1t  = (unsigned short*)carve((size_t)DENC * DENC * 2);  /* [768x768]^T */
  unsigned short* W2t  = (unsigned short*)carve((size_t)DENC * DDEC * 2);  /* [512x768]   */
  unsigned short* W3t  = (unsigned short*)carve((size_t)DDEC * PDIM * 2);  /* [768x512]   */
  unsigned short* A1   = (unsigned short*)carve((size_t)MROWS * PDIM * 2);
  unsigned short* E    = (unsigned short*)carve((size_t)MROWS * DENC * 2);
  float*          LP   = (float*)         carve((size_t)MROWS * DDEC * 4);
  unsigned short* Kd   = (unsigned short*)carve((size_t)MROWS * DDEC * 2);
  unsigned short* MRow = (unsigned short*)carve((size_t)MASKM * DDEC * 2);
  float*          MPred= (float*)         carve((size_t)MASKM * PDIM * 4);
  int*            keepl= (int*)           carve((size_t)MROWS * 4);
  int*            pos  = (int*)           carve((size_t)MROWS * 4);
  int*            sel  = (int*)           carve((size_t)BATCH * LPAT * 4);

  /* weights -> bf16, transposed to [N x K] */
  mae_cvt_t_bf16<<<(DENC * DENC + 255) / 256, 256, 0, stream>>>(W_enc, W1t, DENC, DENC);
  mae_cvt_t_bf16<<<(DENC * DDEC + 255) / 256, 256, 0, stream>>>(W_dec, W2t, DENC, DDEC);
  mae_cvt_t_bf16<<<(DDEC * PDIM + 255) / 256, 256, 0, stream>>>(W_out, W3t, DDEC, PDIM);

  /* mask rows, index maps (+mask output, +cls rows of E), patch gather */
  mae_maskrow<<<(MASKM * DDEC) / 256, 256, 0, stream>>>(mask_tok, dec_pos, MRow);
  mae_indices<<<BATCH, 256, 0, stream>>>(ids, cls_tok, keepl, pos, sel, mask_out, E);
  mae_gather<<<MROWS, 256, 0, stream>>>(images, ids, A1);

  /* GEMM1: kept patches @ W_enc -> E (bf16, +b_enc+enc_pos) */
  mae_gemm<<<dim3(DENC / 64, MROWS / 128), 256, 0, stream>>>(
      A1, W1t, MROWS, DENC, DENC, 0, b_enc, keepl, enc_pos, E, nullptr);

  /* GEMM2: E @ W_dec -> LP (f32, +b_dec) */
  mae_gemm<<<dim3(DDEC / 64, MROWS / 128), 256, 0, stream>>>(
      E, W2t, MROWS, DDEC, DENC, 1, b_dec, nullptr, nullptr, nullptr, LP);

  /* decoder input rows for kept positions */
  mae_build_kd<<<MROWS, 256, 0, stream>>>(LP, dec_pos, pos, Kd);

  /* GEMM3a: (mask_token+dec_pos) @ W_out -> maskpred (batch-shared) */
  mae_gemm<<<dim3(PDIM / 64, MASKM / 128), 256, 0, stream>>>(
      MRow, W3t, MASKM, PDIM, DDEC, 2, b_out, nullptr, nullptr, nullptr, MPred);

  /* GEMM3b: Kd @ W_out -> scatter directly into pred at kept positions */
  mae_gemm<<<dim3(PDIM / 64, MROWS / 128), 256, 0, stream>>>(
      Kd, W3t, MROWS, PDIM, DDEC, 3, b_out, pos, nullptr, nullptr, pred);

  /* broadcast mask-row predictions to the other 146 positions per batch */
  mae_fill_masked<<<BATCH * LPAT, 256, 0, stream>>>(sel, MPred, pred);
}
